// MoELayer_77472620085642
// MI455X (gfx1250) — matrix operand order repaired and checked by reading, not compiled, not run on hardware
//
#include <hip/hip_runtime.h>

// ---------------- problem constants ----------------
#define Hdim 2048
#define Idim 5632
#define Edim 8
#define Tdim 8192               // B*S tokens
#define MAXROWS 16896           // 2*T + 8*63 padding, rounded to /64
#define MAXTILES64 (MAXROWS / 64)
#define ASTRIDE 40              // LDS A-tile row stride (elems): conflict-free ds_load_b128
#define USTRIDE 132             // LDS up-stash row stride (floats): disjoint banks per half-wave

typedef unsigned short u16;
typedef unsigned int   u32;
typedef __attribute__((ext_vector_type(16))) __bf16 v16bf;
typedef __attribute__((ext_vector_type(8)))  float  v8f;

struct alignas(8) U16x4 { u16 x, y, z, w; };

// fp32 -> bf16 round-to-nearest-even
__device__ __forceinline__ u16 f2bf(float f) {
    u32 u = __builtin_bit_cast(u32, f);
    u32 r = (u + 0x7FFFu + ((u >> 16) & 1u)) >> 16;
    return (u16)r;
}

// Build a 16x bf16 WMMA fragment from two 16-byte aligned chunks (global or LDS).
__device__ __forceinline__ v16bf ld_frag(const u16* p0, const u16* p1) {
    union { uint4 q[2]; v16bf v; } u;
    u.q[0] = *(const uint4*)p0;
    u.q[1] = *(const uint4*)p1;
    return u.v;
}

__device__ __forceinline__ v8f wmma_bf16(v16bf a, v16bf b, v8f c) {
    return __builtin_amdgcn_wmma_f32_16x16x32_bf16(
        /*neg_a=*/false, a, /*neg_b=*/false, b,
        /*c_mod=*/(short)0, c, /*reuse_a=*/false, /*reuse_b=*/false);
}

// ---------------- 1) init: zero out, rowmap=-1, counts=0 ----------------
__global__ void init_kernel(float* __restrict__ out, int outN,
                            int* __restrict__ rowmap, int rmN,
                            int* __restrict__ counts) {
    int i = blockIdx.x * blockDim.x + threadIdx.x;
    int stride = gridDim.x * blockDim.x;
    for (int j = i; j < outN; j += stride) out[j] = 0.f;
    for (int j = i; j < rmN; j += stride) rowmap[j] = -1;
    if (i < Edim) counts[i] = 0;
}

// ---------------- 2) weight fp32 -> bf16 (one pass) ----------------
__global__ void convw_kernel(const float* __restrict__ wg, const float* __restrict__ wu,
                             const float* __restrict__ wd,
                             u16* __restrict__ dg, u16* __restrict__ du, u16* __restrict__ dd,
                             size_t quarter /* = E*I*H/4 */) {
    size_t i = (size_t)blockIdx.x * blockDim.x + threadIdx.x;
    size_t stride = (size_t)gridDim.x * blockDim.x;
    size_t total = 3u * quarter;
    for (size_t j = i; j < total; j += stride) {
        const float* s; u16* d; size_t off;
        if (j < quarter)            { s = wg; d = dg; off = j; }
        else if (j < 2 * quarter)   { s = wu; d = du; off = j - quarter; }
        else                        { s = wd; d = dd; off = j - 2 * quarter; }
        float4 f = ((const float4*)s)[off];
        U16x4 o; o.x = f2bf(f.x); o.y = f2bf(f.y); o.z = f2bf(f.z); o.w = f2bf(f.w);
        ((U16x4*)d)[off] = o;
    }
}

// ---------------- 3) routing: logits, softmax, top-2 ----------------
__global__ void route_kernel(const float* __restrict__ x, const float* __restrict__ gw,
                             int* __restrict__ sel_e, float* __restrict__ sel_w,
                             int* __restrict__ counts) {
    int t = blockIdx.x * blockDim.x + threadIdx.x;
    if (t >= Tdim) return;
    float acc[Edim];
#pragma unroll
    for (int e = 0; e < Edim; e++) acc[e] = 0.f;
    const float* xr = x + (size_t)t * Hdim;
    for (int h = 0; h < Hdim; h += 4) {
        float4 xv = *(const float4*)(xr + h);
#pragma unroll
        for (int e = 0; e < Edim; e++) {
            const float* g = gw + (size_t)e * Hdim + h;
            acc[e] += xv.x * g[0] + xv.y * g[1] + xv.z * g[2] + xv.w * g[3];
        }
    }
    float mx = acc[0];
#pragma unroll
    for (int e = 1; e < Edim; e++) mx = fmaxf(mx, acc[e]);
    float p[Edim];
#pragma unroll
    for (int e = 0; e < Edim; e++) p[e] = __expf(acc[e] - mx);
    int e0 = 0; float p0 = p[0];
#pragma unroll
    for (int e = 1; e < Edim; e++) if (p[e] > p0) { p0 = p[e]; e0 = e; }
    int e1 = -1; float p1 = -1.f;
#pragma unroll
    for (int e = 0; e < Edim; e++) if (e != e0 && p[e] > p1) { p1 = p[e]; e1 = e; }
    float inv = __builtin_amdgcn_rcpf(p0 + p1);
    sel_e[2 * t] = e0; sel_e[2 * t + 1] = e1;
    sel_w[2 * t] = p0 * inv; sel_w[2 * t + 1] = p1 * inv;
    atomicAdd(&counts[e0], 1);
    atomicAdd(&counts[e1], 1);
}

// ---------------- 4) scan: 64-aligned expert bases ----------------
__global__ void scan_kernel(const int* __restrict__ counts, int* __restrict__ basep,
                            int* __restrict__ cursor, int* __restrict__ ptot) {
    int acc = 0;
    for (int e = 0; e < Edim; e++) {
        basep[e] = acc; cursor[e] = acc;
        acc += (counts[e] + 63) & ~63;
    }
    basep[Edim] = acc;
    *ptot = acc;
}

// ---------------- 5) assign: compact + gather x to bf16 ----------------
__global__ void assign_kernel(const float* __restrict__ x,
                              const int* __restrict__ sel_e, const float* __restrict__ sel_w,
                              int* __restrict__ cursor, int* __restrict__ rowmap,
                              float* __restrict__ roww, u16* __restrict__ Xc) {
    int t = blockIdx.x;
    __shared__ int s_slot[2];
    if (threadIdx.x == 0) {
#pragma unroll
        for (int j = 0; j < 2; j++) {
            int e = sel_e[2 * t + j];
            int slot = atomicAdd(&cursor[e], 1);
            rowmap[slot] = t;
            roww[slot] = sel_w[2 * t + j];
            s_slot[j] = slot;
        }
    }
    __syncthreads();
    int s0 = s_slot[0], s1 = s_slot[1];
    const float* xr = x + (size_t)t * Hdim;
    for (int h = threadIdx.x; h < Hdim; h += blockDim.x) {
        u16 b = f2bf(xr[h]);
        Xc[(size_t)s0 * Hdim + h] = b;
        Xc[(size_t)s1 * Hdim + h] = b;
    }
}

// ---------------- 6) gate/up GEMM + silu, bf16 WMMA ----------------
// grid (I/128, MAXTILES64), block 256 (8 waves).
// Block tile: M=64 rows x N=128 cols, BOTH matrices.
//   waves 0-3: gate, cols (wave&3)*32   waves 4-7: up, cols (wave&3)*32
// A tile (64x32 bf16) double-buffered through LDS, register-pipelined stage:
//   issue next-chunk global load FIRST, compute, ds_store at end (load long landed),
//   last K-step peeled so the loop body is branch-free.
__global__ __launch_bounds__(256) void ffn_gateup_kernel(
    const u16* __restrict__ Xc, const u16* __restrict__ Wg, const u16* __restrict__ Wu,
    const int* __restrict__ basep, const int* __restrict__ ptot,
    u16* __restrict__ Hb) {
    __shared__ u16   s_a[2][64 * ASTRIDE];
    __shared__ float s_up[64 * USTRIDE];

    int row0 = blockIdx.y * 64;
    if (row0 >= *ptot) return;
    int e = 0;
#pragma unroll
    for (int i = 1; i < Edim; i++) if (row0 >= basep[i]) e = i;

    int tid = threadIdx.x;
    int wave = tid >> 5, lane = tid & 31;
    int mat = wave >> 2;            // 0 = gate, 1 = up
    int colblk = wave & 3;
    int col0 = blockIdx.x * 128 + colblk * 32;
    const u16* w = (mat ? Wu : Wg) + (size_t)e * Idim * Hdim;

    int m   = lane & 15;            // A row / B column within 16-tile
    int kba = (lane >> 4) * 8;      // A fragment K base
    int kbb = (lane >> 4) * 8;      // B fragment K base

    // cooperative A stage: 256 threads x 16B cover 64 rows x 32 elems
    int srow = tid >> 2;
    int schunk = (tid & 3) * 8;
    int soff = srow * ASTRIDE + schunk;
    const u16* sbase = Xc + (size_t)(row0 + srow) * Hdim + schunk;

    const v8f zero = {0.f, 0.f, 0.f, 0.f, 0.f, 0.f, 0.f, 0.f};
    v8f acc[4][2];
#pragma unroll
    for (int rs = 0; rs < 4; rs++)
#pragma unroll
        for (int c = 0; c < 2; c++) acc[rs][c] = zero;

    auto kstep = [&](int cur, int k0) {
        v16bf a[4];
#pragma unroll
        for (int rs = 0; rs < 4; rs++) {
            const u16* ap = &s_a[cur][(rs * 16 + m) * ASTRIDE + kba];
            a[rs] = ld_frag(ap, ap + 16);
        }
#pragma unroll
        for (int c = 0; c < 2; c++) {
            const u16* bp = w + (size_t)(col0 + c * 16 + m) * Hdim + k0 + kbb;
            v16bf b = ld_frag(bp, bp + 16);
#pragma unroll
            for (int rs = 0; rs < 4; rs++)
                acc[rs][c] = wmma_bf16(a[rs], b, acc[rs][c]);
        }
    };

    // prologue: stage k=0
    *(uint4*)&s_a[0][soff] = *(const uint4*)(sbase);
    __syncthreads();

    for (int k0 = 0; k0 < Hdim - 32; k0 += 32) {
        int cur = (k0 >> 5) & 1;
        uint4 stg = *(const uint4*)(sbase + k0 + 32);   // issue stage load first
        __builtin_prefetch(w + (size_t)(col0 + m) * Hdim + k0 + 32 + kbb, 0, 1);
        kstep(cur, k0);                                 // WMMAs hide stage latency
        *(uint4*)&s_a[cur ^ 1][soff] = stg;             // load completed in-order by now
        __syncthreads();
    }
    kstep(((Hdim - 32) >> 5) & 1, Hdim - 32);           // peeled last K-step

    // C/D layout: VGPR r, lane L -> M = r + 8*(L/16), N = L%16
    int n = lane & 15;
    if (mat == 1) {                 // up waves: stash fp32 results in LDS
#pragma unroll
        for (int rs = 0; rs < 4; rs++)
#pragma unroll
            for (int r = 0; r < 8; r++) {
                int mm = rs * 16 + r + ((lane >> 4) << 3);
#pragma unroll
                for (int c = 0; c < 2; c++)
                    s_up[mm * USTRIDE + colblk * 32 + c * 16 + n] = acc[rs][c][r];
            }
    }
    __syncthreads();
    if (mat == 0) {                 // gate waves: silu(g) * u -> bf16
#pragma unroll
        for (int rs = 0; rs < 4; rs++)
#pragma unroll
            for (int r = 0; r < 8; r++) {
                int mm = rs * 16 + r + ((lane >> 4) << 3);
                size_t hb = (size_t)(row0 + mm) * Idim + col0 + n;
#pragma unroll
                for (int c = 0; c < 2; c++) {
                    float g = acc[rs][c][r];
                    float u = s_up[mm * USTRIDE + colblk * 32 + c * 16 + n];
                    float sig = __builtin_amdgcn_rcpf(1.f + __expf(-g));   // v_rcp_f32
                    Hb[hb + c * 16] = f2bf(g * sig * u);
                }
            }
    }
}

// ---------------- 7) down GEMM + weighted scatter ----------------
// grid (H/256, MAXTILES64), block 256 (8 waves). Block tile: M=64, N=256, K=I.
// Wave tile: M=64 x N=32; A tile register-pipelined through LDS, shared by all waves.
__global__ __launch_bounds__(256) void ffn_down_kernel(
    const u16* __restrict__ Hb, const u16* __restrict__ Wd,
    const int* __restrict__ basep, const int* __restrict__ ptot,
    const int* __restrict__ rowmap, const float* __restrict__ roww,
    float* __restrict__ out) {
    __shared__ u16 s_a[2][64 * ASTRIDE];

    int row0 = blockIdx.y * 64;
    if (row0 >= *ptot) return;
    int e = 0;
#pragma unroll
    for (int i = 1; i < Edim; i++) if (row0 >= basep[i]) e = i;

    int tid = threadIdx.x;
    int wave = tid >> 5, lane = tid & 31;
    int col0 = blockIdx.x * 256 + wave * 32;
    const u16* wd = Wd + (size_t)e * Hdim * Idim;

    int m   = lane & 15;
    int kba = (lane >> 4) * 8;
    int kbb = (lane >> 4) * 8;

    int srow = tid >> 2;
    int schunk = (tid & 3) * 8;
    int soff = srow * ASTRIDE + schunk;
    const u16* sbase = Hb + (size_t)(row0 + srow) * Idim + schunk;

    const v8f zero = {0.f, 0.f, 0.f, 0.f, 0.f, 0.f, 0.f, 0.f};
    v8f acc[4][2];
#pragma unroll
    for (int rs = 0; rs < 4; rs++)
#pragma unroll
        for (int c = 0; c < 2; c++) acc[rs][c] = zero;

    auto kstep = [&](int cur, int k0) {
        v16bf a[4];
#pragma unroll
        for (int rs = 0; rs < 4; rs++) {
            const u16* ap = &s_a[cur][(rs * 16 + m) * ASTRIDE + kba];
            a[rs] = ld_frag(ap, ap + 16);
        }
#pragma unroll
        for (int c = 0; c < 2; c++) {
            const u16* bp = wd + (size_t)(col0 + c * 16 + m) * Idim + k0 + kbb;
            v16bf b = ld_frag(bp, bp + 16);
#pragma unroll
            for (int rs = 0; rs < 4; rs++)
                acc[rs][c] = wmma_bf16(a[rs], b, acc[rs][c]);
        }
    };

    *(uint4*)&s_a[0][soff] = *(const uint4*)(sbase);
    __syncthreads();

    for (int k0 = 0; k0 < Idim - 32; k0 += 32) {
        int cur = (k0 >> 5) & 1;
        uint4 stg = *(const uint4*)(sbase + k0 + 32);   // issue stage load first
        __builtin_prefetch(wd + (size_t)(col0 + m) * Idim + k0 + 32 + kbb, 0, 1);
        kstep(cur, k0);
        *(uint4*)&s_a[cur ^ 1][soff] = stg;
        __syncthreads();
    }
    kstep(((Idim - 32) >> 5) & 1, Idim - 32);           // peeled last K-step

    // scatter: out[token] += routing_w * row (exactly 2 atomic adds per element -> deterministic)
    int n = lane & 15;
#pragma unroll
    for (int rs = 0; rs < 4; rs++)
#pragma unroll
        for (int r = 0; r < 8; r++) {
            int row = row0 + rs * 16 + r + ((lane >> 4) << 3);
            int tok = rowmap[row];
            if (tok < 0) continue;            // padded row: never scattered
            float wgt = roww[row];
            size_t ob = (size_t)tok * Hdim + col0 + n;
#pragma unroll
            for (int c = 0; c < 2; c++)
                atomicAdd(&out[ob + c * 16], acc[rs][c][r] * wgt);
        }
}

// ---------------- launch ----------------
extern "C" void kernel_launch(void* const* d_in, const int* in_sizes, int n_in,
                              void* d_out, int out_size, void* d_ws, size_t ws_size,
                              hipStream_t stream) {
    const float* x      = (const float*)d_in[0];
    const float* gate_w = (const float*)d_in[1];
    const float* w_gate = (const float*)d_in[2];
    const float* w_up   = (const float*)d_in[3];
    const float* w_down = (const float*)d_in[4];
    float* out = (float*)d_out;

    char* p = (char*)d_ws;
    auto alloc = [&](size_t bytes) -> char* {
        char* r = p;
        p += (bytes + 255) & ~(size_t)255;
        return r;
    };
    const size_t EIH = (size_t)Edim * Idim * Hdim;
    int*   counts = (int*)alloc(Edim * 4);
    int*   basep  = (int*)alloc((Edim + 1) * 4);
    int*   cursor = (int*)alloc(Edim * 4);
    int*   ptot   = (int*)alloc(4);
    int*   sel_e  = (int*)alloc(2 * Tdim * 4);
    float* sel_w  = (float*)alloc(2 * Tdim * 4);
    int*   rowmap = (int*)alloc(MAXROWS * 4);
    float* roww   = (float*)alloc(MAXROWS * 4);
    u16*   Xc     = (u16*)alloc((size_t)MAXROWS * Hdim * 2);
    u16*   Hb     = (u16*)alloc((size_t)MAXROWS * Idim * 2);
    u16*   Wgb    = (u16*)alloc(EIH * 2);
    u16*   Wub    = (u16*)alloc(EIH * 2);
    u16*   Wdb    = (u16*)alloc(EIH * 2);

    init_kernel<<<4096, 256, 0, stream>>>(out, Tdim * Hdim, rowmap, MAXROWS, counts);
    convw_kernel<<<8192, 256, 0, stream>>>(w_gate, w_up, w_down, Wgb, Wub, Wdb, EIH / 4);
    route_kernel<<<Tdim / 256, 256, 0, stream>>>(x, gate_w, sel_e, sel_w, counts);
    scan_kernel<<<1, 1, 0, stream>>>(counts, basep, cursor, ptot);
    assign_kernel<<<Tdim, 256, 0, stream>>>(x, sel_e, sel_w, cursor, rowmap, roww, Xc);
    ffn_gateup_kernel<<<dim3(Idim / 128, MAXTILES64), 256, 0, stream>>>(Xc, Wgb, Wub, basep, ptot, Hb);
    ffn_down_kernel<<<dim3(Hdim / 256, MAXTILES64), 256, 0, stream>>>(Hb, Wdb, basep, ptot, rowmap, roww, out);
}
